// GraphAttention_73160472920272
// MI455X (gfx1250) — matrix operand order repaired and checked by reading, not compiled, not run on hardware
//
#include <hip/hip_runtime.h>
#include <hip/hip_bf16.h>

// Problem constants (match reference)
#define NODES 512
#define SEQ   64
#define DIM   128
#define HEADS 4
#define KSZ   32
#define EDGES 4096
#define HKDIM 128     // H*K
#define DFF   512     // WF*D
#define HDDIM 512     // H*D
#define NS    (NODES*SEQ)          // 32768 rows
#define SD    (SEQ*DIM)            // 8192
#define ESD   (EDGES*SEQ*DIM)      // 33554432

typedef __attribute__((ext_vector_type(16))) __bf16 v16bf;
typedef __attribute__((ext_vector_type(8)))  __bf16 v8bf;
typedef __attribute__((ext_vector_type(8)))  float  v8f;

static __device__ inline v16bf mk16(v8bf lo, v8bf hi) {
  v16bf r;
#pragma unroll
  for (int i = 0; i < 8; i++) { r[i] = lo[i]; r[i + 8] = hi[i]; }
  return r;
}

static __device__ inline v8f wmma_bf16(v16bf a, v16bf b, v8f c) {
  // D = A(16x32 bf16) * B(32x16 bf16) + C(f32)
  return __builtin_amdgcn_wmma_f32_16x16x32_bf16(false, a, false, b, (short)0, c,
                                                 false, false);
}

static __device__ inline float gelu_t(float x) {
  float x3 = x * x * x;
  float t = tanhf(0.7978845608028654f * (x + 0.044715f * x3));
  return 0.5f * x * (1.0f + t);
}

// ---------------- weight conversion / packing ----------------
__global__ void cvt_bf16_k(const float* __restrict__ src, __bf16* __restrict__ dst,
                           int rows, int cols, int dld, int doff) {
  int i = blockIdx.x * blockDim.x + threadIdx.x;
  if (i >= rows * cols) return;
  int r = i / cols, c = i % cols;
  dst[(size_t)r * dld + doff + c] = (__bf16)src[i];
}

__global__ void copyf_k(const float* __restrict__ src, float* __restrict__ dst, int n) {
  int i = blockIdx.x * blockDim.x + threadIdx.x;
  if (i < n) dst[i] = src[i];
}

// ---------------- LayerNorm (1 wave per row of 128) ----------------
__global__ __launch_bounds__(256) void ln_k(const float* __restrict__ x,
                                            const float* __restrict__ add,
                                            const float* __restrict__ sc,
                                            const float* __restrict__ of,
                                            float* __restrict__ outf,
                                            __bf16* __restrict__ outb, int rows) {
  int wave = blockIdx.x * (blockDim.x >> 5) + (threadIdx.x >> 5);
  int lane = threadIdx.x & 31;
  if (wave >= rows) return;
  size_t base = (size_t)wave * DIM;
  float v[4];
#pragma unroll
  for (int i = 0; i < 4; i++) {
    int c = lane * 4 + i;
    v[i] = x[base + c];
    if (add) v[i] += add[base + c];
  }
  float s = v[0] + v[1] + v[2] + v[3];
  float ss = v[0] * v[0] + v[1] * v[1] + v[2] * v[2] + v[3] * v[3];
#pragma unroll
  for (int d = 16; d >= 1; d >>= 1) {
    s += __shfl_xor(s, d);
    ss += __shfl_xor(ss, d);
  }
  float mu = s * (1.0f / DIM);
  float var = ss * (1.0f / DIM) - mu * mu;
  float rs = rsqrtf(var + 1e-5f);
#pragma unroll
  for (int i = 0; i < 4; i++) {
    int c = lane * 4 + i;
    float y = (v[i] - mu) * rs * sc[c] + of[c];
    if (outf) outf[base + c] = y;
    if (outb) outb[base + c] = (__bf16)y;
  }
}

// ---------------- generic bf16 WMMA GEMM: C = act(A@B + bias) (+resid) ------
__global__ __launch_bounds__(256) void gemm_bf16_k(
    const __bf16* __restrict__ A, const __bf16* __restrict__ B,
    const float* __restrict__ bias, const float* __restrict__ resid,
    float* __restrict__ Cf, __bf16* __restrict__ Cb, int M, int K, int N,
    int act_gelu) {
  int wave = blockIdx.x * (blockDim.x >> 5) + (threadIdx.x >> 5);
  int lane = threadIdx.x & 31;
  int tilesN = N >> 4;
  int tilesM = M >> 4;
  if (wave >= tilesM * tilesN) return;
  int tm = (wave / tilesN) << 4;
  int tn = (wave % tilesN) << 4;
  int half = lane >> 4, mr = lane & 15;
  v8f acc = {};
  const __bf16* arow = A + (size_t)(tm + mr) * K;
  for (int k0 = 0; k0 < K; k0 += 32) {
    v8bf alo = *(const v8bf*)(arow + k0 + half * 8);
    v8bf ahi = *(const v8bf*)(arow + k0 + 16 + half * 8);
    v16bf af = mk16(alo, ahi);
    v16bf bf;
#pragma unroll
    for (int s = 0; s < 16; s++) bf[s] = B[(size_t)(k0 + (s >> 3) * 16 + half * 8 + (s & 7)) * N + tn + mr];
    acc = wmma_bf16(af, bf, acc);
  }
  int col = tn + mr;
  float bv = bias ? bias[col] : 0.0f;
#pragma unroll
  for (int j = 0; j < 8; j++) {
    int row = tm + half * 8 + j;
    float v = acc[j] + bv;
    if (act_gelu) v = gelu_t(v);
    size_t idx = (size_t)row * N + col;
    if (resid) v += resid[idx];
    if (Cf) Cf[idx] = v;
    if (Cb) Cb[idx] = (__bf16)v;
  }
}

// ---------------- fused attention per (node, head) ----------------
__global__ __launch_bounds__(128) void attn_k(const __bf16* __restrict__ qkv,
                                              const float* __restrict__ alibi,
                                              __bf16* __restrict__ attn_out) {
  int n = blockIdx.x >> 2;
  int h = blockIdx.x & 3;
  int tid = threadIdx.x;
  int wv = tid >> 5, lane = tid & 31;
  __shared__ __attribute__((aligned(32))) __bf16 qs[SEQ * KSZ];
  __shared__ __attribute__((aligned(32))) __bf16 ks[SEQ * KSZ];
  __shared__ __attribute__((aligned(32))) __bf16 vs[SEQ * KSZ];
  __shared__ __attribute__((aligned(32))) float probs[SEQ * SEQ];
  __shared__ __attribute__((aligned(32))) __bf16 wbf[SEQ * SEQ];

  for (int i = tid; i < SEQ * KSZ; i += 128) {
    int t = i >> 5, kk = i & 31;
    size_t base = ((size_t)(n * SEQ + t)) * 384 + h * KSZ + kk;
    qs[i] = qkv[base];
    ks[i] = qkv[base + 128];
    vs[i] = qkv[base + 256];
  }
  __syncthreads();

  int half = lane >> 4, mr = lane & 15;
  int tm = wv * 16;
  const float scale = 0.17677669529663687f;  // 1/sqrt(32)

  // A fragment: q rows tm..tm+15, K=32 (single WMMA k-step)
  v16bf af;
  {
    const __bf16* qrow = qs + (tm + mr) * KSZ;
    v8bf alo = *(const v8bf*)(qrow + half * 8);
    v8bf ahi = *(const v8bf*)(qrow + 16 + half * 8);
    af = mk16(alo, ahi);
  }
  for (int Tt = 0; Tt < 4; Tt++) {
    v16bf bf;
#pragma unroll
    for (int j = 0; j < 1; j++) { const __bf16* krow = ks + (Tt * 16 + mr) * KSZ; bf = mk16(*(const v8bf*)(krow + half * 8), *(const v8bf*)(krow + 16 + half * 8)); }  // k^T
    v8f acc = {};
    acc = wmma_bf16(af, bf, acc);
#pragma unroll
    for (int j = 0; j < 8; j++) {
      int t = tm + half * 8 + j;
      int T = Tt * 16 + mr;
      float lg = acc[j] * scale + alibi[((h * SEQ) + t) * SEQ + T];
      if (T > t) lg = -1e30f;
      probs[t * SEQ + T] = lg;
    }
  }
  __syncthreads();

  if (tid < SEQ) {
    int t = tid;
    float m = -3.0e38f;
    for (int T = 0; T < SEQ; T++) m = fmaxf(m, probs[t * SEQ + T]);
    float s = 0.0f;
    for (int T = 0; T < SEQ; T++) {
      float e = __expf(probs[t * SEQ + T] - m);
      probs[t * SEQ + T] = e;
      s += e;
    }
    float inv = 1.0f / s;
    for (int T = 0; T < SEQ; T++) wbf[t * SEQ + T] = (__bf16)(probs[t * SEQ + T] * inv);
  }
  __syncthreads();

  // attn = w @ v  (K=64 -> two k-steps; N=32 -> two n-tiles)
  const __bf16* wrow = wbf + (tm + mr) * SEQ;
  for (int nt = 0; nt < 2; nt++) {
    v8f acc = {};
#pragma unroll
    for (int k0 = 0; k0 < 64; k0 += 32) {
      v8bf alo = *(const v8bf*)(wrow + k0 + half * 8);
      v8bf ahi = *(const v8bf*)(wrow + k0 + 16 + half * 8);
      v16bf a2 = mk16(alo, ahi);
      v16bf b2;
#pragma unroll
      for (int s = 0; s < 16; s++) b2[s] = vs[(k0 + (s >> 3) * 16 + half * 8 + (s & 7)) * KSZ + nt * 16 + mr];
      acc = wmma_bf16(a2, b2, acc);
    }
#pragma unroll
    for (int j = 0; j < 8; j++) {
      int t = tm + half * 8 + j;
      attn_out[((size_t)(n * SEQ + t)) * HKDIM + h * KSZ + nt * 16 + mr] =
          (__bf16)acc[j];
    }
  }
}

// ---------------- deterministic CSR build over receivers ----------------
__global__ void deg_k(const int* __restrict__ rcv, int* __restrict__ deg) {
  int n = blockIdx.x * blockDim.x + threadIdx.x;
  if (n >= NODES) return;
  int c = 0;
  for (int e = 0; e < EDGES; e++) c += (rcv[e] == n);
  deg[n] = c;
}

__global__ void scan_k(const int* __restrict__ deg, int* __restrict__ off) {
  if (blockIdx.x == 0 && threadIdx.x == 0) {
    int a = 0;
    off[0] = 0;
    for (int n = 0; n < NODES; n++) { a += deg[n]; off[n + 1] = a; }
  }
}

__global__ void rank_scatter_k(const int* __restrict__ rcv, const int* __restrict__ off,
                               int* __restrict__ elist) {
  int e = blockIdx.x * blockDim.x + threadIdx.x;
  if (e >= EDGES) return;
  int rv = rcv[e];
  int rk = 0;
  for (int x = 0; x < e; x++) rk += (rcv[x] == rv);
  elist[off[rv] + rk] = e;
}

// ---------------- edge logits (no atomics) ----------------
__global__ void sl_build_k(const float* __restrict__ Am, const float* __restrict__ Bm,
                           const int* __restrict__ snd, const int* __restrict__ rcv,
                           float* __restrict__ sl) {
  int i = blockIdx.x * blockDim.x + threadIdx.x;
  if (i >= ESD) return;
  int e = i >> 13;       // / (S*D)
  int r = i & (SD - 1);
  sl[i] = Am[(size_t)snd[e] * SD + r] + Bm[(size_t)rcv[e] * SD + r];
}

// segment softmax by CSR gather: thread per (node, s*D+d)
__global__ void seg_softmax_k(const float* __restrict__ sl, const int* __restrict__ elist,
                              const int* __restrict__ off, __bf16* __restrict__ aw) {
  int idx = blockIdx.x * blockDim.x + threadIdx.x;
  if (idx >= NODES * SD) return;
  int n = idx >> 13;
  int r = idx & (SD - 1);
  int b = off[n], en = off[n + 1];
  float m = -3.0e38f;
  for (int i = b; i < en; i++) m = fmaxf(m, sl[(size_t)elist[i] * SD + r]);
  float s = 0.0f;
  for (int i = b; i < en; i++) s += __expf(sl[(size_t)elist[i] * SD + r] - m);
  float inv = 1.0f / s;
  for (int i = b; i < en; i++) {
    size_t o = (size_t)elist[i] * SD + r;
    aw[o] = (__bf16)(__expf(sl[o] - m) * inv);
  }
}

// -------- fused: sq = aw@w_sq+b_sq ; msgs = mean_h(sq*sent) (plain stores) ---
__global__ __launch_bounds__(256) void msg_gemm_k(
    const __bf16* __restrict__ aw, const __bf16* __restrict__ wsq,
    const float* __restrict__ bsq, const float* __restrict__ nodes2,
    const int* __restrict__ snd, float* __restrict__ msgs) {
  int wave = blockIdx.x * 8 + (threadIdx.x >> 5);
  int lane = threadIdx.x & 31;
  int tmi = wave >> 3;   // row tile (16 rows of E*S)
  int dt = wave & 7;     // d tile (16 of 128)
  int tm = tmi << 4;
  int half = lane >> 4, mr = lane & 15;

  const __bf16* arow = aw + (size_t)(tm + mr) * DIM;
  v16bf afr[4];
#pragma unroll
  for (int ksb = 0; ksb < 4; ksb++) {
    v8bf alo = *(const v8bf*)(arow + ksb * 32 + half * 8);
    v8bf ahi = *(const v8bf*)(arow + ksb * 32 + 16 + half * 8);
    afr[ksb] = mk16(alo, ahi);
  }
  float msum[8] = {};
#pragma unroll
  for (int h = 0; h < 4; h++) {
    int tn = h * DIM + dt * 16;
    v8f acc = {};
#pragma unroll
    for (int ksb = 0; ksb < 4; ksb++) {
      v16bf bf;
#pragma unroll
      for (int s = 0; s < 16; s++) bf[s] = wsq[(size_t)(ksb * 32 + (s >> 3) * 16 + half * 8 + (s & 7)) * HDDIM + tn + mr];
      acc = wmma_bf16(afr[ksb], bf, acc);
    }
    float bv = bsq[tn + mr];
#pragma unroll
    for (int j = 0; j < 8; j++) msum[j] += acc[j] + bv;
  }
  int d = dt * 16 + mr;
#pragma unroll
  for (int j = 0; j < 8; j++) {
    int row = tm + half * 8 + j;        // row = e*SEQ + s
    int e = row >> 6, s = row & 63;
    float sent = nodes2[((size_t)snd[e] * SEQ + s) * DIM + d];
    msgs[(size_t)row * DIM + d] = msum[j] * sent * 0.25f;
  }
}

// final segment-sum by CSR gather: out[n,s,d] = sum over incoming edges
__global__ void seg_reduce_k(const float* __restrict__ msgs, const int* __restrict__ elist,
                             const int* __restrict__ off, float* __restrict__ out) {
  int idx = blockIdx.x * blockDim.x + threadIdx.x;
  if (idx >= NODES * SD) return;
  int n = idx >> 13;
  int r = idx & (SD - 1);
  int b = off[n], en = off[n + 1];
  float s = 0.0f;
  for (int i = b; i < en; i++) s += msgs[(size_t)elist[i] * SD + r];
  out[idx] = s;
}

// ---------------- host launcher ----------------
extern "C" void kernel_launch(void* const* d_in, const int* in_sizes, int n_in,
                              void* d_out, int out_size, void* d_ws, size_t ws_size,
                              hipStream_t stream) {
  const float* nodes0 = (const float*)d_in[0];
  const int* senders = (const int*)d_in[1];
  const int* receivers = (const int*)d_in[2];
  const float* alibi = (const float*)d_in[3];
  const float* ln1s = (const float*)d_in[4];
  const float* ln1b = (const float*)d_in[5];
  const float* wq = (const float*)d_in[6];
  const float* bq = (const float*)d_in[7];
  const float* wk = (const float*)d_in[8];
  const float* bk = (const float*)d_in[9];
  const float* wvv = (const float*)d_in[10];
  const float* bv = (const float*)d_in[11];
  const float* wo = (const float*)d_in[12];
  const float* bo = (const float*)d_in[13];
  const float* ln2s = (const float*)d_in[14];
  const float* ln2b = (const float*)d_in[15];
  const float* w1 = (const float*)d_in[16];
  const float* b1 = (const float*)d_in[17];
  const float* w2 = (const float*)d_in[18];
  const float* b2 = (const float*)d_in[19];
  const float* ln3s = (const float*)d_in[20];
  const float* ln3b = (const float*)d_in[21];
  const float* w_sa = (const float*)d_in[22];
  const float* b_sa = (const float*)d_in[23];
  const float* w_sq = (const float*)d_in[24];
  const float* b_sq = (const float*)d_in[25];
  float* out = (float*)d_out;

  char* p = (char*)d_ws;
  auto alloc = [&](size_t bytes) -> void* {
    void* r = (void*)p;
    p += (bytes + 255) & ~(size_t)255;
    return r;
  };

  __bf16* wqkv_b = (__bf16*)alloc((size_t)DIM * 384 * 2);
  float* bqkv = (float*)alloc(384 * 4);
  __bf16* wo_b = (__bf16*)alloc((size_t)HKDIM * DIM * 2);
  __bf16* w1_b = (__bf16*)alloc((size_t)DIM * DFF * 2);
  __bf16* w2_b = (__bf16*)alloc((size_t)DFF * DIM * 2);
  __bf16* wsaA_b = (__bf16*)alloc((size_t)DIM * DIM * 2);
  __bf16* wsaB_b = (__bf16*)alloc((size_t)DIM * DIM * 2);
  __bf16* wsq_b = (__bf16*)alloc((size_t)DIM * HDDIM * 2);

  int* deg = (int*)alloc(NODES * 4);
  int* off = (int*)alloc((NODES + 1) * 4);
  int* elist = (int*)alloc(EDGES * 4);

  __bf16* xn_b = (__bf16*)alloc((size_t)NS * DIM * 2);
  __bf16* qkv = (__bf16*)alloc((size_t)NS * 384 * 2);
  __bf16* attn_b = (__bf16*)alloc((size_t)NS * HKDIM * 2);
  float* nodes1 = (float*)alloc((size_t)NS * DIM * 4);
  __bf16* x2_b = (__bf16*)alloc((size_t)NS * DIM * 2);
  __bf16* h_b = (__bf16*)alloc((size_t)NS * DFF * 2);
  float* dense = (float*)alloc((size_t)NS * DIM * 4);
  float* nodes2 = (float*)alloc((size_t)NS * DIM * 4);
  __bf16* nodes2_b = (__bf16*)alloc((size_t)NS * DIM * 2);
  float* Amsg = (float*)alloc((size_t)NS * DIM * 4);
  float* Bmsg = (float*)alloc((size_t)NS * DIM * 4);
  float* sl = (float*)alloc((size_t)ESD * 4);   // reused as msgs after softmax
  __bf16* aw_b = (__bf16*)alloc((size_t)ESD * 2);
  float* msgs = sl;  // sl is dead once aw_b is built; alias to save 134MB

  auto ceil_div = [](int a, int b) { return (a + b - 1) / b; };

  // --- pack weights to bf16 ---
  cvt_bf16_k<<<ceil_div(DIM * HKDIM, 256), 256, 0, stream>>>(wq, wqkv_b, DIM, HKDIM, 384, 0);
  cvt_bf16_k<<<ceil_div(DIM * HKDIM, 256), 256, 0, stream>>>(wk, wqkv_b, DIM, HKDIM, 384, 128);
  cvt_bf16_k<<<ceil_div(DIM * HKDIM, 256), 256, 0, stream>>>(wvv, wqkv_b, DIM, HKDIM, 384, 256);
  copyf_k<<<1, 128, 0, stream>>>(bq, bqkv, 128);
  copyf_k<<<1, 128, 0, stream>>>(bk, bqkv + 128, 128);
  copyf_k<<<1, 128, 0, stream>>>(bv, bqkv + 256, 128);
  cvt_bf16_k<<<ceil_div(HKDIM * DIM, 256), 256, 0, stream>>>(wo, wo_b, HKDIM, DIM, DIM, 0);
  cvt_bf16_k<<<ceil_div(DIM * DFF, 256), 256, 0, stream>>>(w1, w1_b, DIM, DFF, DFF, 0);
  cvt_bf16_k<<<ceil_div(DFF * DIM, 256), 256, 0, stream>>>(w2, w2_b, DFF, DIM, DIM, 0);
  cvt_bf16_k<<<ceil_div(DIM * DIM, 256), 256, 0, stream>>>(w_sa, wsaA_b, DIM, DIM, DIM, 0);
  cvt_bf16_k<<<ceil_div(DIM * DIM, 256), 256, 0, stream>>>(w_sa + DIM * DIM, wsaB_b, DIM, DIM, DIM, 0);
  cvt_bf16_k<<<ceil_div(DIM * HDDIM, 256), 256, 0, stream>>>(w_sq, wsq_b, DIM, HDDIM, HDDIM, 0);

  // --- deterministic CSR over receivers (overlaps with transformer work) ---
  deg_k<<<ceil_div(NODES, 256), 256, 0, stream>>>(receivers, deg);
  scan_k<<<1, 1, 0, stream>>>(deg, off);
  rank_scatter_k<<<ceil_div(EDGES, 256), 256, 0, stream>>>(receivers, off, elist);

  // --- LN1 -> xn (bf16) ---
  ln_k<<<NS / 8, 256, 0, stream>>>(nodes0, nullptr, ln1s, ln1b, nullptr, xn_b, NS);

  // --- fused QKV GEMM: [NS,128] @ [128,384] ---
  {
    int waves = (NS / 16) * (384 / 16);
    gemm_bf16_k<<<ceil_div(waves, 8), 256, 0, stream>>>(
        xn_b, wqkv_b, bqkv, nullptr, nullptr, qkv, NS, DIM, 384, 0);
  }

  // --- attention per (node, head) ---
  attn_k<<<NODES * HEADS, 128, 0, stream>>>(qkv, alibi, attn_b);

  // --- out-proj + residual: nodes1 = nodes0 + attn @ wo + bo ---
  {
    int waves = (NS / 16) * (DIM / 16);
    gemm_bf16_k<<<ceil_div(waves, 8), 256, 0, stream>>>(
        attn_b, wo_b, bo, nodes0, nodes1, nullptr, NS, HKDIM, DIM, 0);
  }

  // --- LN2 -> x2 (bf16) ---
  ln_k<<<NS / 8, 256, 0, stream>>>(nodes1, nullptr, ln2s, ln2b, nullptr, x2_b, NS);

  // --- FFN: h = gelu(x2 @ w1 + b1) (bf16) ; dense = h @ w2 + b2 (f32) ---
  {
    int waves = (NS / 16) * (DFF / 16);
    gemm_bf16_k<<<ceil_div(waves, 8), 256, 0, stream>>>(
        x2_b, w1_b, b1, nullptr, nullptr, h_b, NS, DIM, DFF, 1);
  }
  {
    int waves = (NS / 16) * (DIM / 16);
    gemm_bf16_k<<<ceil_div(waves, 8), 256, 0, stream>>>(
        h_b, w2_b, b2, nullptr, dense, nullptr, NS, DFF, DIM, 0);
  }

  // --- LN3: nodes2 = LN(nodes1 + dense) (f32 + bf16) ---
  ln_k<<<NS / 8, 256, 0, stream>>>(nodes1, dense, ln3s, ln3b, nodes2, nodes2_b, NS);

  // --- per-node projections for edge attention logits ---
  {
    int waves = (NS / 16) * (DIM / 16);
    gemm_bf16_k<<<ceil_div(waves, 8), 256, 0, stream>>>(
        nodes2_b, wsaA_b, b_sa, nullptr, Amsg, nullptr, NS, DIM, DIM, 0);
    gemm_bf16_k<<<ceil_div(waves, 8), 256, 0, stream>>>(
        nodes2_b, wsaB_b, nullptr, nullptr, Bmsg, nullptr, NS, DIM, DIM, 0);
  }

  // --- edge logits + CSR-gather segment softmax (atomic-free) ---
  sl_build_k<<<ceil_div(ESD, 256), 256, 0, stream>>>(Amsg, Bmsg, senders, receivers, sl);
  seg_softmax_k<<<ceil_div(NODES * SD, 256), 256, 0, stream>>>(sl, elist, off, aw_b);

  // --- fused message GEMM + head-mean (plain stores into msgs == sl) ---
  {
    int waves = ((EDGES * SEQ) / 16) * (DIM / 16);  // 16384 * 8
    msg_gemm_k<<<waves / 8, 256, 0, stream>>>(aw_b, wsq_b, b_sq, nodes2, senders, msgs);
  }

  // --- final CSR-gather segment sum into out ---
  seg_reduce_k<<<ceil_div(NODES * SD, 256), 256, 0, stream>>>(msgs, elist, off, out);
}
